// graphtransformer_input_layer_mb_32212254720357
// MI455X (gfx1250) — matrix operand order repaired and checked by reading, not compiled, not run on hardware
//
#include <hip/hip_runtime.h>
#include <hip/hip_bf16.h>

typedef __attribute__((ext_vector_type(16))) _Float16 v16h;
typedef __attribute__((ext_vector_type(8)))  float    v8f;
typedef __attribute__((ext_vector_type(4)))  unsigned v4u;
typedef __attribute__((ext_vector_type(8)))  int      v8i;
typedef __attribute__((ext_vector_type(4)))  int      v4i;

#define N_SRC 200000
#define N_DST 100000
#define E_EDGES 400000
#define INF_DIM 128
#define NHEAD 8
#define HDIM 32
#define HD 256            // NHEAD*HDIM
#define TM 128            // GEMM M tile (8 waves x 16 rows)
#define ATTN_SCALE 0.17677669529663687f   // 1/sqrt(32)

// ---------- float <-> order-preserving uint encoding for atomic max ----------
__device__ __forceinline__ unsigned fenc(float f) {
    unsigned b = __float_as_uint(f);
    return (b & 0x80000000u) ? ~b : (b | 0x80000000u);
}
__device__ __forceinline__ float fdec(unsigned u) {
    return __uint_as_float((u & 0x80000000u) ? (u ^ 0x80000000u) : ~u);
}
#define NEG_INF_ENC 0x007FFFFFu   // fenc(-inf)

// ---------------------------------------------------------------------------
// Generic fill
// ---------------------------------------------------------------------------
__global__ void fill_u32(unsigned* __restrict__ p, unsigned val, long n) {
    long i = (long)blockIdx.x * blockDim.x + threadIdx.x;
    if (i < n) p[i] = val;
}

// ---------------------------------------------------------------------------
// f32 [N,K] row-major  ->  f16 [N,Kpad] row-major, zero-padded K
// ---------------------------------------------------------------------------
__global__ void cvt_pad_f16(const float* __restrict__ src, _Float16* __restrict__ dst,
                            int N, int K, int Kpad) {
    long i = (long)blockIdx.x * blockDim.x + threadIdx.x;
    if (i >= (long)N * Kpad) return;
    int n = (int)(i / Kpad), k = (int)(i - (long)n * Kpad);
    dst[i] = (_Float16)((k < K) ? src[(long)n * K + k] : 0.f);
}

// ---------------------------------------------------------------------------
// WMMA GEMM with fused epilogues.
//   C[M,Nout] = A[M,K] @ Wh[Nout,Kpad]^T   (f16 operands, f32 accumulate)
//   Wh is pre-converted f16, zero-padded to Kpad; staged to LDS by the TDM.
// modes:
//   0 SRC   : + bias[col] + rowExtra[row,col] + colExtra[col]     -> Cout
//   1 DST   : + bias[col] + colExtra[col]                         -> Cout
//   2 HADD  : + bias[col] + gTab[gIdx[row], col]                  -> Cout
//   3 PLAIN : + bias[col]                                         -> Cout
//   4 REDNA : atomicAdd(red[row*8 + col/32], v * wna[col&31])     (no Cout)
// 256 threads (8 wave32), one 128-row M tile per block, full Nout per block.
// ---------------------------------------------------------------------------
__global__ void gemm_ep(const float* __restrict__ A, const _Float16* __restrict__ Wh,
                        const float* __restrict__ bias,
                        const float* __restrict__ rowExtra,
                        const float* __restrict__ colExtra,
                        const int*   __restrict__ gIdx,
                        const float* __restrict__ gTab,
                        const float* __restrict__ wna,
                        float* __restrict__ Cout,
                        float* __restrict__ red,
                        int M, int K, int Kpad, int Nout, int mode)
{
    extern __shared__ _Float16 lds[];
    _Float16* la = lds;                       // [TM][Kpad]
    _Float16* lw = lds + (size_t)TM * Kpad;   // [Nout][Kpad]

    const int tid   = threadIdx.x;
    const int tileM = blockIdx.x * TM;

#if defined(__gfx1250__)
    // ---- Tensor Data Mover: DMA the whole f16 weight panel into LDS --------
    // 2D descriptor: tile == tensor == [Nout rows][Kpad halves], data_size=2B.
    if (tid == 0) {
        unsigned long long ga = (unsigned long long)Wh;
        unsigned ldsoff = (unsigned)((size_t)TM * Kpad * sizeof(_Float16)); // lw offset
        v4u g0 = { 1u,                                  // count=1, user desc
                   ldsoff,                              // lds_addr [63:32]
                   (unsigned)ga,                        // global_addr lo
                   ((unsigned)(ga >> 32) & 0x01FFFFFFu) | 0x80000000u }; // addr hi | type=2
        v8i g1 = { 1 << 16,                             // data_size=1 (2 bytes)
                   (Kpad & 0xFFFF) << 16,               // tensor_dim0 lo16
                   (Nout & 0xFFFF) << 16,               // tensor_dim0 hi=0 | tensor_dim1 lo16
                   (Kpad & 0xFFFF) << 16,               // tensor_dim1 hi=0 | tile_dim0
                   Nout & 0xFFFF,                       // tile_dim1 | tile_dim2=0
                   Kpad,                                // tensor_dim0_stride lo32
                   0, 0 };
        v4i z = { 0, 0, 0, 0 };
#if __clang_major__ >= 23
        v8i z8 = { 0, 0, 0, 0, 0, 0, 0, 0 };
        __builtin_amdgcn_tensor_load_to_lds(g0, g1, z, z, z8, 0);
#else
        __builtin_amdgcn_tensor_load_to_lds(g0, g1, z, z, 0);
#endif
        __builtin_amdgcn_s_wait_tensorcnt(0);
    }
#else
    for (int i = tid; i < Nout * Kpad; i += 256) lw[i] = Wh[i];
#endif

    // Stage A tile (f32 -> f16, zero-pad K) into LDS
    const int atot = TM * Kpad;
    for (int i = tid; i < atot; i += 256) {
        int r = i / Kpad, k = i - r * Kpad;
        int row = tileM + r;
        float v = 0.f;
        if (row < M && k < K) v = A[(long)row * K + k];
        la[i] = (_Float16)v;
    }
    __syncthreads();

    const int lane = tid & 31;
    const int wave = tid >> 5;
    const int g    = lane >> 4;     // half-wave group
    const int mr   = lane & 15;
    const int nkc  = Kpad >> 5;     // K chunks of 32
    const int nnb  = Nout >> 4;     // N blocks of 16

    for (int nb = 0; nb < nnb; ++nb) {
        v8f c = {};
        for (int kc = 0; kc < nkc; ++kc) {
            v16h a, b;
            // A 16x32 f16 fragment: lane holds row (wave*16+mr);
            // half pair v -> K = kc*32 + 8*g + 2*(v&3) + 16*(v>>2)
            const _Float16* ap = &la[(size_t)(wave * 16 + mr) * Kpad + kc * 32 + 8 * g];
#pragma unroll
            for (int v = 0; v < 8; ++v) {
                int kk = 2 * (v & 3) + 16 * (v >> 2);
                a[2 * v]     = ap[kk];
                a[2 * v + 1] = ap[kk + 1];
            }
            // B 32x16 f16 fragment: lane holds col (nb*16+mr); half j -> K = kc*32 + 8*g + (j&7) + 16*(j>>3)
            const _Float16* bp = &lw[(size_t)(nb * 16 + mr) * Kpad + kc * 32 + 8 * g];
#pragma unroll
            for (int j = 0; j < 16; ++j) b[j] = bp[(j & 7) + 16 * (j >> 3)];

            c = __builtin_amdgcn_wmma_f32_16x16x32_f16(false, a, false, b,
                                                       (short)0, c, false, false);
        }
        // Epilogue: element r of C frag -> row = r + 8*g, col = mr (within tile)
        const int col = nb * 16 + mr;
        const float bb = bias ? bias[col] : 0.f;
        const float ce = colExtra ? colExtra[col] : 0.f;
#pragma unroll
        for (int r = 0; r < 8; ++r) {
            int row = tileM + wave * 16 + r + 8 * g;
            if (row >= M) continue;
            float v = c[r] + bb + ce;
            if (mode == 0) {
                v += rowExtra[(long)row * Nout + col];
                Cout[(long)row * Nout + col] = v;
            } else if (mode == 1) {
                Cout[(long)row * Nout + col] = v;
            } else if (mode == 2) {
                v += gTab[(long)gIdx[row] * Nout + col];
                Cout[(long)row * Nout + col] = v;
            } else if (mode == 3) {
                Cout[(long)row * Nout + col] = v;
            } else {
                atomicAdd(&red[(long)row * NHEAD + (col >> 5)], v * wna[col & 31]);
            }
        }
    }
}

// ---------------------------------------------------------------------------
// Segment-softmax passes (channel softmax over hadd[E,128], head softmax over
// logits[E,8]); max via ordered-uint atomicMax, sum via f32 atomicAdd.
// ---------------------------------------------------------------------------
__global__ void edge_max_h(const float* __restrict__ hadd, const int* __restrict__ dst,
                           unsigned* __restrict__ hmax, long total) {
    long i = (long)blockIdx.x * blockDim.x + threadIdx.x;
    if (i >= total) return;
    long e = i >> 7; int c = (int)(i & 127);
    atomicMax(&hmax[(long)dst[e] * INF_DIM + c], fenc(hadd[i]));
}

__global__ void edge_sum_h(const float* __restrict__ hadd, const int* __restrict__ dst,
                           const unsigned* __restrict__ hmax, float* __restrict__ hsum,
                           long total) {
    long i = (long)blockIdx.x * blockDim.x + threadIdx.x;
    if (i >= total) return;
    long e = i >> 7; int c = (int)(i & 127);
    long si = (long)dst[e] * INF_DIM + c;
    atomicAdd(&hsum[si], __expf(hadd[i] - fdec(hmax[si])));
}

__global__ void edge_max_l(const float* __restrict__ ak, const float* __restrict__ aq,
                           const int* __restrict__ src, const int* __restrict__ dst,
                           unsigned* __restrict__ lmax, long total) {
    long i = (long)blockIdx.x * blockDim.x + threadIdx.x;
    if (i >= total) return;
    long e = i >> 3; int h = (int)(i & 7);
    float lg = (ak[(long)src[e] * NHEAD + h] + aq[(long)dst[e] * NHEAD + h]) * ATTN_SCALE;
    atomicMax(&lmax[(long)dst[e] * NHEAD + h], fenc(lg));
}

__global__ void edge_sum_l(const float* __restrict__ ak, const float* __restrict__ aq,
                           const int* __restrict__ src, const int* __restrict__ dst,
                           const unsigned* __restrict__ lmax, float* __restrict__ lsum,
                           long total) {
    long i = (long)blockIdx.x * blockDim.x + threadIdx.x;
    if (i >= total) return;
    long e = i >> 3; int h = (int)(i & 7);
    long si = (long)dst[e] * NHEAD + h;
    float lg = (ak[(long)src[e] * NHEAD + h] + aq[si]) * ATTN_SCALE;
    atomicAdd(&lsum[si], __expf(lg - fdec(lmax[si])));
}

// ---------------------------------------------------------------------------
// Final edge kernel: wave32 per edge.
// ---------------------------------------------------------------------------
__global__ void edge_final(const float* __restrict__ hadd,
                           const unsigned* __restrict__ hmax, const float* __restrict__ hsum,
                           const float* __restrict__ aq, const float* __restrict__ ak,
                           const unsigned* __restrict__ lmax, const float* __restrict__ lsum,
                           const float* __restrict__ vmat,
                           const int* __restrict__ src, const int* __restrict__ dst,
                           const float* __restrict__ W_eattn, const float* __restrict__ b_eattn,
                           const float* __restrict__ W_merge, const float* __restrict__ b_merge,
                           float* __restrict__ agg, int nE)
{
    __shared__ float sWe[NHEAD * INF_DIM];
    __shared__ float sbe[NHEAD];
    __shared__ float sWm[NHEAD * 16];
    __shared__ float sbm[NHEAD];
    int tid = threadIdx.x;
    for (int i = tid; i < NHEAD * INF_DIM; i += 256) sWe[i] = W_eattn[i];
    if (tid < NHEAD * 16) sWm[tid] = W_merge[tid];
    if (tid < NHEAD) { sbe[tid] = b_eattn[tid]; sbm[tid] = b_merge[tid]; }
    __syncthreads();

    int wave = tid >> 5, lane = tid & 31;
    long e = (long)blockIdx.x * 8 + wave;
    if (e >= nE) return;
    int s = src[e], d = dst[e];
    __builtin_prefetch(&vmat[(long)s * HD], 0, 0);

    // channel softmax values for this lane's 4 channels
    float t[4];
#pragma unroll
    for (int i = 0; i < 4; ++i) {
        int c = lane * 4 + i;
        long si = (long)d * INF_DIM + c;
        t[i] = __expf(hadd[e * INF_DIM + c] - fdec(hmax[si])) / hsum[si];
    }
    float ta[NHEAD], sa[NHEAD], mg[NHEAD];
#pragma unroll
    for (int h = 0; h < NHEAD; ++h) {
        float p = 0.f;
#pragma unroll
        for (int i = 0; i < 4; ++i) p += t[i] * sWe[h * INF_DIM + lane * 4 + i];
#pragma unroll
        for (int o = 16; o > 0; o >>= 1) p += __shfl_xor(p, o, 32);
        ta[h] = p + sbe[h];
    }
#pragma unroll
    for (int h = 0; h < NHEAD; ++h) {
        long si = (long)d * NHEAD + h;
        float lg = (ak[(long)s * NHEAD + h] + aq[si]) * ATTN_SCALE;
        sa[h] = __expf(lg - fdec(lmax[si])) / lsum[si];
    }
#pragma unroll
    for (int h = 0; h < NHEAD; ++h) {
        float m = sbm[h];
#pragma unroll
        for (int j = 0; j < NHEAD; ++j)
            m += sWm[h * 16 + j] * sa[j] + sWm[h * 16 + 8 + j] * ta[j];
        mg[h] = m;
    }
#pragma unroll
    for (int i = 0; i < 8; ++i) {
        int c = i * 32 + lane;   // head = i
        atomicAdd(&agg[(long)d * HD + c], vmat[(long)s * HD + c] * mg[i]);
    }
}

// ---------------------------------------------------------------------------
// Row-wise LayerNorm over 256 channels, in place on d_out. Block = row.
// ---------------------------------------------------------------------------
__global__ void layernorm256(float* __restrict__ out, const float* __restrict__ gamma,
                             const float* __restrict__ beta) {
    __shared__ float rs[8], rs2[8];
    int tid = threadIdx.x;
    long row = blockIdx.x;
    float x = out[row * HD + tid];
    float s = x, s2 = x * x;
#pragma unroll
    for (int o = 16; o > 0; o >>= 1) { s += __shfl_xor(s, o, 32); s2 += __shfl_xor(s2, o, 32); }
    int wave = tid >> 5, lane = tid & 31;
    if (lane == 0) { rs[wave] = s; rs2[wave] = s2; }
    __syncthreads();
    if (tid == 0) {
        float a = 0.f, b = 0.f;
#pragma unroll
        for (int i = 0; i < 8; ++i) { a += rs[i]; b += rs2[i]; }
        rs[0] = a; rs2[0] = b;
    }
    __syncthreads();
    float mu  = rs[0]  * (1.f / HD);
    float var = rs2[0] * (1.f / HD) - mu * mu;
    out[row * HD + tid] = (x - mu) * rsqrtf(var + 1e-5f) * gamma[tid] + beta[tid];
}

// ---------------------------------------------------------------------------
extern "C" void kernel_launch(void* const* d_in, const int* in_sizes, int n_in,
                              void* d_out, int out_size, void* d_ws, size_t ws_size,
                              hipStream_t stream) {
    const float* x_user        = (const float*)d_in[0];
    const float* x_item        = (const float*)d_in[1];
    const float* node_emb_user = (const float*)d_in[2];
    const float* edge_emb      = (const float*)d_in[3];
    const float* edge_feats    = (const float*)d_in[4];
    const int*   src_idx       = (const int*)d_in[5];
    const int*   dst_idx       = (const int*)d_in[6];
    const float* W_nf_user     = (const float*)d_in[7];
    const float* b_nf_user     = (const float*)d_in[8];
    const float* W_nf_item     = (const float*)d_in[9];
    const float* b_nf_item     = (const float*)d_in[10];
    const float* W_ef          = (const float*)d_in[11];
    const float* b_ef          = (const float*)d_in[12];
    const float* W_eattn       = (const float*)d_in[13];
    const float* b_eattn       = (const float*)d_in[14];
    const float* W_merge       = (const float*)d_in[15];
    const float* b_merge       = (const float*)d_in[16];
    const float* W_q_item      = (const float*)d_in[17];
    const float* b_q_item      = (const float*)d_in[18];
    const float* W_k_user      = (const float*)d_in[19];
    const float* b_k_user      = (const float*)d_in[20];
    const float* W_v_user      = (const float*)d_in[21];
    const float* b_v_user      = (const float*)d_in[22];
    const float* W_na_user     = (const float*)d_in[23];
    const float* W_na_item     = (const float*)d_in[24];
    const float* ln_gamma      = (const float*)d_in[25];
    const float* ln_beta       = (const float*)d_in[26];

    float* ws = (float*)d_ws;
    size_t o = 0;
    float*    src_feat = ws + o; o += (size_t)N_SRC  * INF_DIM;
    float*    dst_feat = ws + o; o += (size_t)N_DST  * INF_DIM;
    float*    hadd     = ws + o; o += (size_t)E_EDGES* INF_DIM;
    float*    vmat     = ws + o; o += (size_t)N_SRC  * HD;
    float*    aq       = ws + o; o += (size_t)N_DST  * NHEAD;
    float*    ak       = ws + o; o += (size_t)N_SRC  * NHEAD;
    unsigned* hmax     = (unsigned*)(ws + o); o += (size_t)N_DST * INF_DIM;
    float*    hsum     = ws + o; o += (size_t)N_DST * INF_DIM;
    unsigned* lmax     = (unsigned*)(ws + o); o += (size_t)N_DST * NHEAD;
    float*    lsum     = ws + o; o += (size_t)N_DST * NHEAD;
    // f16 padded weight panels (for TDM staging)
    _Float16* wh = (_Float16*)(ws + o);
    size_t ho = 0;
    _Float16* Wh_nf_user = wh + ho; ho += (size_t)128 * 320;
    _Float16* Wh_nf_item = wh + ho; ho += (size_t)128 * 224;
    _Float16* Wh_ef      = wh + ho; ho += (size_t)128 * 64;
    _Float16* Wh_q       = wh + ho; ho += (size_t)256 * 128;
    _Float16* Wh_k       = wh + ho; ho += (size_t)256 * 128;
    _Float16* Wh_v       = wh + ho; ho += (size_t)256 * 128;
    float* out = (float*)d_out;

    auto fill = [&](void* p, unsigned v, long n) {
        fill_u32<<<dim3((unsigned)((n + 255) / 256)), dim3(256), 0, stream>>>((unsigned*)p, v, n);
    };
    fill(hmax, NEG_INF_ENC, (long)N_DST * INF_DIM);
    fill(hsum, 0u,          (long)N_DST * INF_DIM);
    fill(lmax, NEG_INF_ENC, (long)N_DST * NHEAD);
    fill(lsum, 0u,          (long)N_DST * NHEAD);
    fill(aq,   0u,          (long)N_DST * NHEAD);
    fill(ak,   0u,          (long)N_SRC * NHEAD);
    fill(out,  0u,          (long)N_DST * HD);

    auto cvt = [&](const float* src, _Float16* dst, int N, int K) {
        int Kpad = ((K + 31) / 32) * 32;
        long n = (long)N * Kpad;
        cvt_pad_f16<<<dim3((unsigned)((n + 255) / 256)), dim3(256), 0, stream>>>(src, dst, N, K, Kpad);
    };
    cvt(W_nf_user, Wh_nf_user, 128, 300);
    cvt(W_nf_item, Wh_nf_item, 128, 200);
    cvt(W_ef,      Wh_ef,      128, 64);
    cvt(W_q_item,  Wh_q,       256, 128);
    cvt(W_k_user,  Wh_k,       256, 128);
    cvt(W_v_user,  Wh_v,       256, 128);

    auto gemm = [&](const float* A, const _Float16* Wh_, const float* bias,
                    const float* rowExtra, const float* colExtra,
                    const int* gIdx, const float* gTab, const float* wna,
                    float* Cout, float* red, int M, int K, int Nout, int mode) {
        int Kpad = ((K + 31) / 32) * 32;
        size_t sh = (size_t)(TM + Nout) * Kpad * sizeof(_Float16);
        gemm_ep<<<dim3((unsigned)((M + TM - 1) / TM)), dim3(256), sh, stream>>>(
            A, Wh_, bias, rowExtra, colExtra, gIdx, gTab, wna, Cout, red,
            M, K, Kpad, Nout, mode);
    };

    // 1) src_feat = x_user @ W_nf_user^T + b + node_emb_user + edge_emb
    gemm(x_user, Wh_nf_user, b_nf_user, node_emb_user, edge_emb,
         nullptr, nullptr, nullptr, src_feat, nullptr, N_SRC, 300, INF_DIM, 0);
    // 2) dst_feat = x_item @ W_nf_item^T + b + edge_emb
    gemm(x_item, Wh_nf_item, b_nf_item, nullptr, edge_emb,
         nullptr, nullptr, nullptr, dst_feat, nullptr, N_DST, 200, INF_DIM, 1);
    // 3) hadd = edge_feats @ W_ef^T + b + src_feat[src_idx]   (fused gather)
    gemm(edge_feats, Wh_ef, b_ef, nullptr, nullptr,
         src_idx, src_feat, nullptr, hadd, nullptr, E_EDGES, 64, INF_DIM, 2);
    // 4) aq = reduce_na(dst_feat @ W_q^T + b_q, W_na_item)  (Q never stored)
    gemm(dst_feat, Wh_q, b_q_item, nullptr, nullptr,
         nullptr, nullptr, W_na_item, nullptr, aq, N_DST, INF_DIM, HD, 4);
    // 5) ak = reduce_na(src_feat @ W_k^T + b_k, W_na_user)  (K never stored)
    gemm(src_feat, Wh_k, b_k_user, nullptr, nullptr,
         nullptr, nullptr, W_na_user, nullptr, ak, N_SRC, INF_DIM, HD, 4);
    // 6) v = src_feat @ W_v^T + b_v
    gemm(src_feat, Wh_v, b_v_user, nullptr, nullptr,
         nullptr, nullptr, nullptr, vmat, nullptr, N_SRC, INF_DIM, HD, 3);

    // 7) segment-softmax stats
    {
        long tot = (long)E_EDGES * INF_DIM;
        unsigned g = (unsigned)((tot + 255) / 256);
        edge_max_h<<<dim3(g), dim3(256), 0, stream>>>(hadd, dst_idx, hmax, tot);
        edge_sum_h<<<dim3(g), dim3(256), 0, stream>>>(hadd, dst_idx, hmax, hsum, tot);
        long totl = (long)E_EDGES * NHEAD;
        unsigned gl = (unsigned)((totl + 255) / 256);
        edge_max_l<<<dim3(gl), dim3(256), 0, stream>>>(ak, aq, src_idx, dst_idx, lmax, totl);
        edge_sum_l<<<dim3(gl), dim3(256), 0, stream>>>(ak, aq, src_idx, dst_idx, lmax, lsum, totl);
    }

    // 8) per-edge attention merge + scatter into d_out
    edge_final<<<dim3((E_EDGES + 7) / 8), dim3(256), 0, stream>>>(
        hadd, hmax, hsum, aq, ak, lmax, lsum, vmat, src_idx, dst_idx,
        W_eattn, b_eattn, W_merge, b_merge, out, E_EDGES);

    // 9) LayerNorm in place on d_out
    layernorm256<<<dim3(N_DST), dim3(256), 0, stream>>>(out, ln_gamma, ln_beta);
}
